// Model_57183194579227
// MI455X (gfx1250) — matrix operand order repaired and checked by reading, not compiled, not run on hardware
//
#include <hip/hip_runtime.h>
#include <stdint.h>

typedef __attribute__((ext_vector_type(16))) _Float16 v16h;
typedef __attribute__((ext_vector_type(8)))  _Float16 v8h;
typedef __attribute__((ext_vector_type(8)))  float    v8f;
typedef __attribute__((ext_vector_type(4)))  unsigned int v4u;
typedef __attribute__((ext_vector_type(8)))  int      v8i_;
typedef __attribute__((ext_vector_type(4)))  int      v4i_;

#define B_   8192
#define N_   160
#define NB_  12

// ---------------------------------------------------------------------------
// TDM: DMA a contiguous [16 x 160] f16 tile from global into LDS.
// D# per CDNA5 ISA: 2D tensor, data_size=2B,
// tensor_dim0=160, tile=160x16, dim0 stride=160. Issued once per wave,
// tracked on TENSORcnt.
// ---------------------------------------------------------------------------
__device__ __forceinline__ void tdm_load_tile_16x160(const _Float16* gsrc,
                                                     _Float16* lds_dst) {
#if __has_builtin(__builtin_amdgcn_tensor_load_to_lds)
    unsigned long long ga = (unsigned long long)(uintptr_t)gsrc;
    unsigned int lo = (unsigned int)(uintptr_t)lds_dst;   // LDS byte offset
    v4u g0;
    g0.x = 1u;                                            // count=1 (valid user D#)
    g0.y = lo;                                            // lds_addr
    g0.z = (unsigned int)(ga & 0xffffffffu);              // global_addr[31:0]
    g0.w = (unsigned int)((ga >> 32) & 0x1ffffffu)        // global_addr[56:32]
         | (2u << 30);                                    // type = 2 ("image")
    v8i_ g1;
    g1[0] = (int)(1u << 16);       // workgroup_mask=0, data_size=1 (2 bytes)
    g1[1] = (int)(160u << 16);     // tensor_dim0 = 160 (bits [79:48] low half)
    g1[2] = (int)(16u  << 16);     // tensor_dim1 = 16  (bits [111:80] low half)
    g1[3] = (int)(160u << 16);     // tile_dim0  = 160  (bits [127:112])
    g1[4] = 16;                    // tile_dim1  = 16   (bits [143:128])
    g1[5] = 160;                   // tensor_dim0_stride = 160 (bits [207:160])
    g1[6] = 0;
    g1[7] = 0;
    v4i_ gz = {0, 0, 0, 0};
#if __clang_major__ >= 23
    v8i_ gz8 = {0, 0, 0, 0, 0, 0, 0, 0};
    __builtin_amdgcn_tensor_load_to_lds(g0, g1, gz, gz, gz8, 0);
#else
    __builtin_amdgcn_tensor_load_to_lds(g0, g1, gz, gz, 0);
#endif
    __builtin_amdgcn_s_wait_tensorcnt(0);
#else
    (void)gsrc; (void)lds_dst;
#endif
}

// ---------------------------------------------------------------------------
// Convert all block weights (qw,kw,vw,pw) fp32 -> f16, layout per block:
// [qw | kw | vw | pw], each 160x160 row-major (out, in).
// ---------------------------------------------------------------------------
__global__ void convert_w_kernel(const float* __restrict__ qw,
                                 const float* __restrict__ kw,
                                 const float* __restrict__ vw,
                                 const float* __restrict__ pw,
                                 _Float16* __restrict__ w16) {
    int idx = blockIdx.x * 256 + threadIdx.x;
    const int total = NB_ * 4 * N_ * N_;
    if (idx >= total) return;
    int m   = idx / (N_ * N_);
    int r   = idx % (N_ * N_);
    int blk = m >> 2;
    int sel = m & 3;
    const float* src = (sel == 0) ? qw : (sel == 1) ? kw : (sel == 2) ? vw : pw;
    w16[idx] = (_Float16)src[blk * (N_ * N_) + r];
}

// ---------------------------------------------------------------------------
// SIREN sin-MLP + sinusoidal time embedding -> emb [B,160] (fp32 + f16 copy)
// ---------------------------------------------------------------------------
__global__ void siren_temb_kernel(const float* __restrict__ x,
                                  const int*   __restrict__ t,
                                  const float* __restrict__ sw0,
                                  const float* __restrict__ sb0,
                                  const float* __restrict__ sw,
                                  const float* __restrict__ sb,
                                  float* __restrict__ emb32,
                                  _Float16* __restrict__ emb16) {
    int b = blockIdx.x * 256 + threadIdx.x;
    if (b >= B_) return;
    float xi[9];
    #pragma unroll
    for (int k = 0; k < 9; ++k) xi[k] = x[b * 9 + k];
    float h[32];
    #pragma unroll
    for (int r = 0; r < 32; ++r) {
        float s = sb0[r];
        #pragma unroll
        for (int k = 0; k < 9; ++k) s += sw0[r * 9 + k] * xi[k];
        h[r] = __sinf(s);
    }
    for (int L = 0; L < 9; ++L) {
        float h2[32];
        #pragma unroll 4
        for (int r = 0; r < 32; ++r) {
            float s = sb[L * 32 + r];
            #pragma unroll
            for (int c = 0; c < 32; ++c) s += sw[L * 1024 + r * 32 + c] * h[c];
            h2[r] = __sinf(s);
        }
        #pragma unroll
        for (int r = 0; r < 32; ++r) h[r] = h2[r];
    }
    #pragma unroll
    for (int r = 0; r < 32; ++r) {
        emb32[b * N_ + r] = h[r];
        emb16[b * N_ + r] = (_Float16)h[r];
    }
    // time embedding: cols [32,96) = sin, [96,160) = cos
    float tf = (float)t[b];
    const float cc = -0.1461958789f; // -ln(10000)/63
    for (int j = 0; j < 64; ++j) {
        float ang = tf * __expf(cc * (float)j);
        float sv = __sinf(ang), cv = __cosf(ang);
        emb32[b * N_ + 32 + j] = sv;  emb16[b * N_ + 32 + j] = (_Float16)sv;
        emb32[b * N_ + 96 + j] = cv;  emb16[b * N_ + 96 + j] = (_Float16)cv;
    }
}

// ---------------------------------------------------------------------------
// Fused q/k/v GEMM: [16 x 160] emb tile  x  W^T (3 matrices, 30 col-tiles)
// via v_wmma_f32_16x16x32_f16, K = 5 steps of 32. 256 thr = 8 waves / block.
// A tile staged to LDS by the Tensor Data Mover.
// ---------------------------------------------------------------------------
__global__ void __launch_bounds__(256)
qkv_wmma_kernel(const _Float16* __restrict__ emb16,
                const _Float16* __restrict__ w16blk,   // qw|kw|vw (f16)
                const float* __restrict__ qb,
                const float* __restrict__ kb,
                const float* __restrict__ vb,
                float* __restrict__ q32,
                float* __restrict__ k32,
                float* __restrict__ v32) {
    __shared__ __align__(32) _Float16 As[16 * N_];
    const int tid  = threadIdx.x;
    const int wave = tid >> 5;
    const int lane = tid & 31;
    const int row0 = blockIdx.x * 16;

#if __has_builtin(__builtin_amdgcn_tensor_load_to_lds)
    if (wave == 0) tdm_load_tile_16x160(emb16 + (size_t)row0 * N_, As);
#else
    {
        const unsigned int* src = (const unsigned int*)(emb16 + (size_t)row0 * N_);
        unsigned int* dst = (unsigned int*)As;
        #pragma unroll
        for (int i = 0; i < 5; ++i) dst[tid + i * 256] = src[tid + i * 256];
    }
#endif
    __syncthreads();

    const int rsel = lane & 15;     // A row / D column selector
    const int hsel = lane >> 4;     // half-wave selector

    // Build the 5 A fragments (ISA 16-bit A 16x32 layout)
    v16h a[5];
    #pragma unroll
    for (int ks = 0; ks < 5; ++ks) {
        v8h lo = *(const v8h*)&As[rsel * N_ + ks * 32 + 8 * hsel];
        v8h hi = *(const v8h*)&As[rsel * N_ + ks * 32 + 8 * hsel + 16];
        a[ks] = __builtin_shufflevector(lo, hi, 0,1,2,3,4,5,6,7,8,9,10,11,12,13,14,15);
    }

    for (int ct = wave; ct < 30; ct += 8) {
        const int sel = ct / 10;           // 0=q 1=k 2=v
        const int c   = ct % 10;           // output col tile
        const _Float16* w = w16blk + sel * (N_ * N_);
        const int n = c * 16 + rsel;       // global output column
        __builtin_prefetch(w + (size_t)n * N_, 0, 0);
        v8f acc = {};
        #pragma unroll
        for (int ks = 0; ks < 5; ++ks) {
            // B frag (32x16): lane holds col n, K halfs 8h..8h+7 and 16+8h..16+8h+7 of W[n,:]
            v8h blo = *(const v8h*)&w[(size_t)n * N_ + ks * 32 + 8 * hsel];
            v8h bhi = *(const v8h*)&w[(size_t)n * N_ + ks * 32 + 8 * hsel + 16];
            v16h bf = __builtin_shufflevector(blo, bhi, 0,1,2,3,4,5,6,7,8,9,10,11,12,13,14,15);
            acc = __builtin_amdgcn_wmma_f32_16x16x32_f16(
                false, a[ks], false, bf, (short)0, acc, false, false);
        }
        const float* bias = (sel == 0) ? qb : (sel == 1) ? kb : vb;
        float* dst        = (sel == 0) ? q32 : (sel == 1) ? k32 : v32;
        const float bn = bias[n];
        #pragma unroll
        for (int r = 0; r < 8; ++r)
            dst[(size_t)(row0 + r + 8 * hsel) * N_ + n] = acc[r] + bn;
    }
}

// ---------------------------------------------------------------------------
// Outer-product softmax attention + residual SiLU.  One wave per batch row.
// k/v rows staged straight into LDS with async global->LDS loads (ASYNCcnt).
// att[i] = sum_j softmax_j(q_i*k_j) * v_j ; embA = silu(emb + att)
// max_j(q_i*k_j) computed analytically from sign(q_i) and kmax/kmin.
// ---------------------------------------------------------------------------
__global__ void __launch_bounds__(256)
attn_silu_kernel(const float* __restrict__ emb32,
                 const float* __restrict__ q32,
                 const float* __restrict__ k32,
                 const float* __restrict__ v32,
                 float* __restrict__ embA32,
                 _Float16* __restrict__ embA16) {
    __shared__ float kbuf[8][N_];
    __shared__ float vbuf[8][N_];
    const int tid  = threadIdx.x;
    const int wave = tid >> 5;
    const int lane = tid & 31;
    const int row  = blockIdx.x * 8 + wave;

    const float* kr = k32 + (size_t)row * N_;
    const float* vr = v32 + (size_t)row * N_;

    // async global->LDS: per-lane 32-bit transfers, 5 per lane per buffer
    #pragma unroll
    for (int jj = 0; jj < N_ / 32; ++jj) {
        const int j = lane + jj * 32;
        unsigned int kds = (unsigned int)(uintptr_t)&kbuf[wave][j];
        unsigned int vds = (unsigned int)(uintptr_t)&vbuf[wave][j];
        unsigned long long ka = (unsigned long long)(uintptr_t)(kr + j);
        unsigned long long va = (unsigned long long)(uintptr_t)(vr + j);
        asm volatile("global_load_async_to_lds_b32 %0, %1, off"
                     :: "v"(kds), "v"(ka) : "memory");
        asm volatile("global_load_async_to_lds_b32 %0, %1, off"
                     :: "v"(vds), "v"(va) : "memory");
    }
    asm volatile("s_wait_asynccnt 0x0" ::: "memory");
    __syncthreads();

    float lmax = -1e30f, lmin = 1e30f;
    #pragma unroll
    for (int jj = 0; jj < N_ / 32; ++jj) {
        float kv = kbuf[wave][lane + jj * 32];
        lmax = fmaxf(lmax, kv);
        lmin = fminf(lmin, kv);
    }
    #pragma unroll
    for (int m = 16; m >= 1; m >>= 1) {
        lmax = fmaxf(lmax, __shfl_xor(lmax, m, 32));
        lmin = fminf(lmin, __shfl_xor(lmin, m, 32));
    }

    const float* qr = q32 + (size_t)row * N_;
    for (int i = lane; i < N_; i += 32) {
        const float qi = qr[i];
        const float mm = (qi >= 0.f) ? qi * lmax : qi * lmin;
        float s = 0.f, d = 0.f;
        #pragma unroll 4
        for (int j = 0; j < N_; ++j) {
            float e = __expf(qi * kbuf[wave][j] - mm);
            s += e;
            d += e * vbuf[wave][j];
        }
        float z = emb32[(size_t)row * N_ + i] + d / s;
        float o = z / (1.f + __expf(-z));            // silu
        embA32[(size_t)row * N_ + i] = o;
        embA16[(size_t)row * N_ + i] = (_Float16)o;
    }
}

// ---------------------------------------------------------------------------
// p-projection GEMM + residual SiLU epilogue:
// emb = silu(embA + embA @ pw^T + pb)
// ---------------------------------------------------------------------------
__global__ void __launch_bounds__(256)
proj_silu_wmma_kernel(const _Float16* __restrict__ embA16,
                      const _Float16* __restrict__ pw16,
                      const float* __restrict__ pb,
                      const float* __restrict__ embA32,
                      float* __restrict__ emb32,
                      _Float16* __restrict__ emb16) {
    __shared__ __align__(32) _Float16 As[16 * N_];
    const int tid  = threadIdx.x;
    const int wave = tid >> 5;
    const int lane = tid & 31;
    const int row0 = blockIdx.x * 16;

#if __has_builtin(__builtin_amdgcn_tensor_load_to_lds)
    if (wave == 0) tdm_load_tile_16x160(embA16 + (size_t)row0 * N_, As);
#else
    {
        const unsigned int* src = (const unsigned int*)(embA16 + (size_t)row0 * N_);
        unsigned int* dst = (unsigned int*)As;
        #pragma unroll
        for (int i = 0; i < 5; ++i) dst[tid + i * 256] = src[tid + i * 256];
    }
#endif
    __syncthreads();

    const int rsel = lane & 15;
    const int hsel = lane >> 4;

    v16h a[5];
    #pragma unroll
    for (int ks = 0; ks < 5; ++ks) {
        v8h lo = *(const v8h*)&As[rsel * N_ + ks * 32 + 8 * hsel];
        v8h hi = *(const v8h*)&As[rsel * N_ + ks * 32 + 8 * hsel + 16];
        a[ks] = __builtin_shufflevector(lo, hi, 0,1,2,3,4,5,6,7,8,9,10,11,12,13,14,15);
    }

    for (int c = wave; c < 10; c += 8) {
        const int n = c * 16 + rsel;
        v8f acc = {};
        #pragma unroll
        for (int ks = 0; ks < 5; ++ks) {
            v8h blo = *(const v8h*)&pw16[(size_t)n * N_ + ks * 32 + 8 * hsel];
            v8h bhi = *(const v8h*)&pw16[(size_t)n * N_ + ks * 32 + 8 * hsel + 16];
            v16h bf = __builtin_shufflevector(blo, bhi, 0,1,2,3,4,5,6,7,8,9,10,11,12,13,14,15);
            acc = __builtin_amdgcn_wmma_f32_16x16x32_f16(
                false, a[ks], false, bf, (short)0, acc, false, false);
        }
        const float bn = pb[n];
        #pragma unroll
        for (int r = 0; r < 8; ++r) {
            const int rr = row0 + r + 8 * hsel;
            float z = embA32[(size_t)rr * N_ + n] + acc[r] + bn;
            float o = z / (1.f + __expf(-z));        // silu
            emb32[(size_t)rr * N_ + n] = o;
            emb16[(size_t)rr * N_ + n] = (_Float16)o;
        }
    }
}

// ---------------------------------------------------------------------------
// Final projection 160 -> 3
// ---------------------------------------------------------------------------
__global__ void out_kernel(const float* __restrict__ emb32,
                           const float* __restrict__ ow,
                           const float* __restrict__ ob,
                           float* __restrict__ out) {
    int b = blockIdx.x * 256 + threadIdx.x;
    if (b >= B_) return;
    #pragma unroll
    for (int o = 0; o < 3; ++o) {
        float s = ob[o];
        #pragma unroll 8
        for (int k = 0; k < N_; ++k) s += emb32[(size_t)b * N_ + k] * ow[o * N_ + k];
        out[b * 3 + o] = s;
    }
}

// ---------------------------------------------------------------------------
extern "C" void kernel_launch(void* const* d_in, const int* in_sizes, int n_in,
                              void* d_out, int out_size, void* d_ws, size_t ws_size,
                              hipStream_t stream) {
    const float* x   = (const float*)d_in[0];
    const int*   t   = (const int*)d_in[1];
    const float* sw0 = (const float*)d_in[2];
    const float* sb0 = (const float*)d_in[3];
    const float* sw  = (const float*)d_in[4];
    const float* sb  = (const float*)d_in[5];
    const float* qw  = (const float*)d_in[6];
    const float* qb  = (const float*)d_in[7];
    const float* kw  = (const float*)d_in[8];
    const float* kb  = (const float*)d_in[9];
    const float* vw  = (const float*)d_in[10];
    const float* vb  = (const float*)d_in[11];
    const float* pw  = (const float*)d_in[12];
    const float* pb  = (const float*)d_in[13];
    const float* ow  = (const float*)d_in[14];
    const float* ob  = (const float*)d_in[15];
    float* out = (float*)d_out;

    // workspace carve-up (256B aligned slices)
    char* base = (char*)d_ws;
    size_t off = 0;
    auto carve = [&](size_t bytes) -> void* {
        off = (off + 255) & ~(size_t)255;
        void* p = base + off;
        off += bytes;
        return p;
    };
    float*    emb32  = (float*)   carve((size_t)B_ * N_ * 4);
    float*    embA32 = (float*)   carve((size_t)B_ * N_ * 4);
    _Float16* emb16  = (_Float16*)carve((size_t)B_ * N_ * 2);
    _Float16* embA16 = (_Float16*)carve((size_t)B_ * N_ * 2);
    float*    q32    = (float*)   carve((size_t)B_ * N_ * 4);
    float*    k32    = (float*)   carve((size_t)B_ * N_ * 4);
    float*    v32    = (float*)   carve((size_t)B_ * N_ * 4);
    _Float16* w16    = (_Float16*)carve((size_t)NB_ * 4 * N_ * N_ * 2);
    (void)ws_size; (void)n_in; (void)in_sizes; (void)out_size;

    const int wtotal = NB_ * 4 * N_ * N_;
    convert_w_kernel<<<(wtotal + 255) / 256, 256, 0, stream>>>(qw, kw, vw, pw, w16);
    siren_temb_kernel<<<B_ / 256, 256, 0, stream>>>(x, t, sw0, sb0, sw, sb, emb32, emb16);

    for (int blk = 0; blk < NB_; ++blk) {
        const _Float16* wblk = w16 + (size_t)blk * 4 * N_ * N_;   // q|k|v|p
        qkv_wmma_kernel<<<B_ / 16, 256, 0, stream>>>(
            emb16, wblk, qb + blk * N_, kb + blk * N_, vb + blk * N_,
            q32, k32, v32);
        attn_silu_kernel<<<B_ / 8, 256, 0, stream>>>(
            emb32, q32, k32, v32, embA32, embA16);
        proj_silu_wmma_kernel<<<B_ / 16, 256, 0, stream>>>(
            embA16, wblk + 3 * N_ * N_, pb + blk * N_, embA32, emb32, emb16);
    }

    out_kernel<<<B_ / 256, 256, 0, stream>>>(emb32, ow, ob, out);
}
